// LeNet5_11845519802954
// MI455X (gfx1250) — matrix operand order repaired and checked by reading, not compiled, not run on hardware
//
#include <hip/hip_runtime.h>
#include <hip/hip_bf16.h>

// ============================================================================
// LeNet-5 forward (B=8192) for MI455X / gfx1250, wave32 + WMMA.
//  - conv1/conv2 (implicit GEMM, channel-padded K' = C*32 so there is no
//    K-tail and no in-loop div/mod) and fc/fc1/fc2 all use
//    v_wmma_f32_16x16x32_f16 (f16 in, f32 accum).
//  - Fragment loads are branch-free and unconditional (float4 /
//    global_load_b128 in the FC path). Edge N-tiles need no masking at all:
//    invalid B columns only produce invalid D columns, which are not stored.
//    K-tails zero only the A fragment (0 * finite = 0).
//  - Every producer fuses the global min/max of its output (needed by the
//    next quant_inject) into its epilogue via monotonic-uint atomics.
//  - Fault injection approximated with a deterministic hash (expected count
//    matched to the reference schedule); threefry is not reproducible here.
// ============================================================================

typedef __attribute__((ext_vector_type(16))) _Float16 v16h;
typedef __attribute__((ext_vector_type(8)))  float    v8f;

// ---- monotonic float<->uint encoding so atomicMin/Max(uint) order floats ----
__device__ __forceinline__ unsigned enc_f32(float f) {
  unsigned u = __float_as_uint(f);
  return (u & 0x80000000u) ? ~u : (u | 0x80000000u);
}
__device__ __forceinline__ float dec_f32(unsigned e) {
  return (e & 0x80000000u) ? __uint_as_float(e ^ 0x80000000u)
                           : __uint_as_float(~e);
}

// Block-level min/max commit (uniform mm pointer; collective)
__device__ __forceinline__ void commit_minmax(float lmn, float lmx, unsigned* mm) {
  __shared__ unsigned smn, smx;
  if (threadIdx.x == 0) { smn = 0xFFFFFFFFu; smx = 0u; }
  __syncthreads();
  atomicMin(&smn, enc_f32(lmn));
  atomicMax(&smx, enc_f32(lmx));
  __syncthreads();
  if (threadIdx.x == 0) { atomicMin(&mm[0], smn); atomicMax(&mm[1], smx); }
}

__device__ __forceinline__ float hash01(unsigned idx, unsigned seed) {
  unsigned h = idx * 2654435761u ^ (seed * 0x9E3779B9u);
  h ^= h >> 16; h *= 0x85EBCA6Bu; h ^= h >> 13; h *= 0xC2B2AE35u; h ^= h >> 16;
  return (float)(h >> 8) * (1.0f / 16777216.0f);
}

__device__ __forceinline__ v8f wmma_f16(const v16h& a, const v16h& b, const v8f& c) {
  return __builtin_amdgcn_wmma_f32_16x16x32_f16(false, a, false, b, (short)0, c,
                                                false, false);
}

// ---------------------------------------------------------------------------
__global__ void init_mm_kernel(unsigned* mm, int slots) {
  int t = blockIdx.x * blockDim.x + threadIdx.x;
  if (t < slots) { mm[2 * t] = 0xFFFFFFFFu; mm[2 * t + 1] = 0u; }
}

// ---------------------------------------------------------------------------
// FC GEMM: Y[M,N] = X[M,K] @ W[N,K]^T + b, one wave per 16x16 tile.
// M % 16 == 0 (M = 8192). K floats are 16B-multiples (400/120/84).
// ---------------------------------------------------------------------------
template <int N, int K>
__global__ void fc_wmma_kernel(const float* __restrict__ X,
                               const float* __restrict__ Wt,
                               const float* __restrict__ bias,
                               float* __restrict__ Y,
                               int M, unsigned* mm_out) {
  constexpr int TILESN = (N + 15) >> 4;
  constexpr int KFULL = K >> 5;
  constexpr int KREM = K & 31;
  int wave = (blockIdx.x * blockDim.x + threadIdx.x) >> 5;
  int lane = threadIdx.x & 31;
  int hs = lane >> 4, lx = lane & 15;
  int tilesM = M >> 4;
  int tm = wave / TILESN, tn = wave % TILESN;
  float lmn = __builtin_inff(), lmx = -__builtin_inff();

  if (tm < tilesM) {                       // wave-uniform guard
    int rowBase = tm << 4, colBase = tn << 4;
    int m = rowBase + lx;
    int n = colBase + lx;
    const float* xr = X + (size_t)m * K;
    const float* wr = Wt + (size_t)(n < N ? n : N - 1) * K;  // clamp: edge cols
    v8f acc = {};                          // produce unused D cols (not stored)

    for (int s = 0; s < KFULL; ++s) {      // unconditional vector loads
      int k0 = s << 5;
      __builtin_prefetch(xr + k0 + 64, 0, 1);   // global_prefetch_b8
      const float4* pa0 = (const float4*)(xr + k0 + hs * 8);
      const float4* pa1 = (const float4*)(xr + k0 + 16 + hs * 8);
      float4 a0 = pa0[0], a1 = pa0[1];
      float4 a2 = pa1[0], a3 = pa1[1];
      const float4* pb0 = (const float4*)(wr + k0 + hs * 8);
      const float4* pb1 = (const float4*)(wr + k0 + 16 + hs * 8);
      float4 b0 = pb0[0], b1 = pb0[1], b2 = pb1[0], b3 = pb1[1];
      v16h a, b;
      a[0]=(_Float16)a0.x;  a[1]=(_Float16)a0.y;  a[2]=(_Float16)a0.z;  a[3]=(_Float16)a0.w;
      a[4]=(_Float16)a1.x;  a[5]=(_Float16)a1.y;  a[6]=(_Float16)a1.z;  a[7]=(_Float16)a1.w;
      a[8]=(_Float16)a2.x;  a[9]=(_Float16)a2.y;  a[10]=(_Float16)a2.z; a[11]=(_Float16)a2.w;
      a[12]=(_Float16)a3.x; a[13]=(_Float16)a3.y; a[14]=(_Float16)a3.z; a[15]=(_Float16)a3.w;
      b[0]=(_Float16)b0.x;  b[1]=(_Float16)b0.y;  b[2]=(_Float16)b0.z;  b[3]=(_Float16)b0.w;
      b[4]=(_Float16)b1.x;  b[5]=(_Float16)b1.y;  b[6]=(_Float16)b1.z;  b[7]=(_Float16)b1.w;
      b[8]=(_Float16)b2.x;  b[9]=(_Float16)b2.y;  b[10]=(_Float16)b2.z; b[11]=(_Float16)b2.w;
      b[12]=(_Float16)b3.x; b[13]=(_Float16)b3.y; b[14]=(_Float16)b3.z; b[15]=(_Float16)b3.w;
      acc = wmma_f16(a, b, acc);
    }
    if constexpr (KREM != 0) {             // tail: clamp loads, zero A only
      constexpr int k0 = KFULL << 5;
      v16h a, b;
      #pragma unroll
      for (int j = 0; j < 8; ++j) {
        int ka = k0 + hs * 8 + j;
        int kb = k0 + 16 + hs * 8 + j;
        float va = xr[ka < K ? ka : 0];
        float vb = xr[kb < K ? kb : 0];
        a[j]     = (ka < K) ? (_Float16)va : (_Float16)0.0f;
        a[8 + j] = (kb < K) ? (_Float16)vb : (_Float16)0.0f;
      }
      #pragma unroll
      for (int j = 0; j < 16; ++j) {       // B garbage at k>=K is nulled by A=0
        int kk = k0 + (j < 8 ? hs * 8 + j : 16 + hs * 8 + (j - 8));
        b[j] = (_Float16)wr[kk < K ? kk : K - 1];
      }
      acc = wmma_f16(a, b, acc);
    }
    #pragma unroll
    for (int r = 0; r < 8; ++r) {          // C/D layout: m = base + hs*8 + r
      int mo = rowBase + hs * 8 + r;
      int no = colBase + lx;
      if (no < N) {
        float v = acc[r] + bias[no];
        Y[(size_t)mo * N + no] = v;
        lmn = fminf(lmn, v); lmx = fmaxf(lmx, v);
      }
    }
  }
  if (mm_out) commit_minmax(lmn, lmx, mm_out);
}

// ---------------------------------------------------------------------------
// Implicit-GEMM 5x5 VALID conv, NCHW, one wave per 16-pixel x OC tile.
// GEMM K-axis is channel-padded: K' = C*32, slot = c*32 + t, t<25 is filter
// element (t/5, t%5), t in [25,32) is zero-padded in A. Each 32-step is one
// channel, so per-lane intra-channel gather offsets are loop-invariant: no
// div/mod in the hot loop and no K-tail.
// ---------------------------------------------------------------------------
template <int C, int H, int W, int OC, int OH, int OW>
__global__ void conv_wmma_kernel(const float* __restrict__ IN,
                                 const float* __restrict__ Wt,   // [OC, C*25]
                                 const float* __restrict__ bias,
                                 float* __restrict__ OUT,        // [Bn,OC,OH,OW]
                                 int Bn, unsigned* mm_out) {
  int M = Bn * OH * OW;                    // multiple of 16 for both convs
  int wave = (blockIdx.x * blockDim.x + threadIdx.x) >> 5;
  int lane = threadIdx.x & 31;
  int hs = lane >> 4, lx = lane & 15;
  int tilesM = M >> 4;
  float lmn = __builtin_inff(), lmx = -__builtin_inff();

  if (wave < tilesM) {
    int rowBase = wave << 4;
    int m = rowBase + lx;
    int px = m % (OH * OW);
    int bi = m / (OH * OW);
    int oy = px / OW, ox = px % OW;
    const float* inb = IN + (size_t)bi * C * H * W + (size_t)oy * W + ox;
    const float* wr = Wt + (size_t)(lx < OC ? lx : OC - 1) * (C * 25);

    // Loop-invariant per-lane gather offsets (one-time div/mod).
    int offA0[8], offA1[8], offB[16];
    bool vA1[8];
    #pragma unroll
    for (int j = 0; j < 8; ++j) {
      int t0 = hs * 8 + j;                 // 0..15: always a real filter elem
      offA0[j] = (t0 / 5) * W + (t0 % 5);
      int t1 = 16 + hs * 8 + j;            // 16..31: padded when >= 25
      vA1[j] = (t1 < 25);
      int t1c = vA1[j] ? t1 : 24;
      offA1[j] = (t1c / 5) * W + (t1c % 5);
    }
    #pragma unroll
    for (int j = 0; j < 16; ++j) {
      int tb = (j < 8) ? hs * 8 + j : 16 + hs * 8 + (j - 8);  // clamp: A=0 nulls padded slots
      offB[j] = tb < 25 ? tb : 24;
    }

    v8f acc = {};
    #pragma unroll
    for (int c = 0; c < C; ++c) {          // one channel per WMMA K-step
      const float* pc = inb + (size_t)c * H * W;
      const float* wc = wr + c * 25;
      v16h a, b;
      #pragma unroll
      for (int j = 0; j < 8; ++j) {
        a[j] = (_Float16)pc[offA0[j]];
        float v1 = pc[offA1[j]];
        a[8 + j] = vA1[j] ? (_Float16)v1 : (_Float16)0.0f;
      }
      #pragma unroll
      for (int j = 0; j < 16; ++j)
        b[j] = (_Float16)wc[offB[j]];
      acc = wmma_f16(a, b, acc);
    }
    #pragma unroll
    for (int r = 0; r < 8; ++r) {
      int mo = rowBase + hs * 8 + r;
      int no = lx;                          // oc
      if (no < OC) {
        int p2 = mo % (OH * OW);
        int b2 = mo / (OH * OW);
        int y2 = p2 / OW, x2 = p2 % OW;
        float v = acc[r] + bias[no];
        OUT[(((size_t)b2 * OC + no) * OH + y2) * OW + x2] = v;
        lmn = fminf(lmn, v); lmx = fmaxf(lmx, v);
      }
    }
  }
  if (mm_out) commit_minmax(lmn, lmx, mm_out);
}

// ---------------------------------------------------------------------------
// 2x2/2 maxpool, NCHW, fused output min/max.
// ---------------------------------------------------------------------------
__global__ void maxpool_kernel(const float* __restrict__ IN, float* __restrict__ OUT,
                               int C, int H, int W, int n_out, unsigned* mm_out) {
  int OH = H >> 1, OW = W >> 1;
  float lmn = __builtin_inff(), lmx = -__builtin_inff();
  int stride = gridDim.x * blockDim.x;
  for (int i = blockIdx.x * blockDim.x + threadIdx.x; i < n_out; i += stride) {
    int ow = i % OW; int t = i / OW;
    int oh = t % OH; t /= OH;
    int c = t % C;  int b = t / C;
    size_t base = (((size_t)b * C + c) * H + (oh << 1)) * W + (ow << 1);
    float v = fmaxf(fmaxf(IN[base], IN[base + 1]),
                    fmaxf(IN[base + W], IN[base + W + 1]));
    OUT[i] = v;
    lmn = fminf(lmn, v); lmx = fmaxf(lmx, v);
  }
  if (mm_out) commit_minmax(lmn, lmx, mm_out);
}

// ---------------------------------------------------------------------------
// quant-dequant in place. Optional ReLU-after-dequant, optional hash faults,
// optional fused min/max of the result (feeds the next quant site).
// ---------------------------------------------------------------------------
__global__ void quant_kernel(float* __restrict__ X, int n,
                             const unsigned* __restrict__ mm_in,
                             unsigned* mm_out,
                             int do_relu, int do_fault, unsigned seed,
                             const int* __restrict__ tc_ptr) {
  float mn = dec_f32(mm_in[0]);
  float mx = dec_f32(mm_in[1]);
  float scale = (mx - mn) * (1.0f / 255.0f);
  float inv = (scale > 0.0f) ? 1.0f / scale : 0.0f;
  float zp = fminf(fmaxf(-mn * inv, 0.0f), 255.0f);
  float p = 0.0f;
  if (do_fault) {
    // per-site faults = tc*8*0.1/4; mean burst size 4.5 -> expected zeroed
    float nf = (float)(*tc_ptr) * 0.2f;
    p = (nf / 4.5f) / (float)n;
  }
  float lmn = __builtin_inff(), lmx = -__builtin_inff();
  int stride = gridDim.x * blockDim.x;
  for (int i = blockIdx.x * blockDim.x + threadIdx.x; i < n; i += stride) {
    float q = fminf(fmaxf(rintf(zp + X[i] * inv), 0.0f), 255.0f);
    if (do_fault && hash01((unsigned)i, seed) < p) q = 0.0f;
    float v = scale * (q - zp);
    if (do_relu) v = fmaxf(v, 0.0f);
    X[i] = v;
    lmn = fminf(lmn, v); lmx = fmaxf(lmx, v);
  }
  if (mm_out) commit_minmax(lmn, lmx, mm_out);
}

// ---------------------------------------------------------------------------
__global__ void logsoftmax_kernel(const float* __restrict__ IN,
                                  float* __restrict__ OUT, int rows, int cols) {
  int r = blockIdx.x * blockDim.x + threadIdx.x;
  if (r >= rows) return;
  const float* p = IN + (size_t)r * cols;
  float m = -__builtin_inff();
  for (int c = 0; c < cols; ++c) m = fmaxf(m, p[c]);
  float s = 0.0f;
  for (int c = 0; c < cols; ++c) s += __expf(p[c] - m);
  float ls = __logf(s);
  for (int c = 0; c < cols; ++c) OUT[(size_t)r * cols + c] = p[c] - m - ls;
}

// ===========================================================================
extern "C" void kernel_launch(void* const* d_in, const int* in_sizes, int n_in,
                              void* d_out, int out_size, void* d_ws, size_t ws_size,
                              hipStream_t stream) {
  const float* x   = (const float*)d_in[0];
  const float* w1  = (const float*)d_in[1];
  const float* b1  = (const float*)d_in[2];
  const float* w2  = (const float*)d_in[3];
  const float* b2  = (const float*)d_in[4];
  const float* fw  = (const float*)d_in[5];
  const float* fb  = (const float*)d_in[6];
  const float* f1w = (const float*)d_in[7];
  const float* f1b = (const float*)d_in[8];
  const float* f2w = (const float*)d_in[9];
  const float* f2b = (const float*)d_in[10];
  const int*   tc  = (const int*)d_in[11];
  float* out = (float*)d_out;

  const int B = 8192;
  const int N_A1 = B * 6 * 28 * 28;   // 38,535,168
  const int N_P1 = B * 6 * 14 * 14;   //  9,633,792
  const int N_A2 = B * 16 * 10 * 10;  // 13,107,200
  const int N_P2 = B * 400;           //  3,276,800
  const int N_F0 = B * 120;
  const int N_F1 = B * 84;
  const int N_LG = B * 10;

  // Workspace layout (A2/F0 alias the dead A1 region, P2 aliases dead P1).
  char* ws = (char*)d_ws;
  unsigned* mm = (unsigned*)ws;                           // 11 (min,max) slots
  float* A1 = (float*)(ws + 256);
  float* P1 = (float*)(ws + 256 + (size_t)N_A1 * 4);
  float* A2 = A1;
  float* P2 = P1;
  float* F0 = A1;
  float* F1 = (float*)(ws + 256 + (4ull << 20));
  float* LG = (float*)(ws + 256 + (8ull << 20));

  unsigned* mmA1  = mm + 0;   // conv1 raw out
  unsigned* mmR1  = mm + 2;   // after quant0+relu (feeds fault quant)
  unsigned* mmP1  = mm + 4;   // pool1 out
  unsigned* mmA2  = mm + 6;
  unsigned* mmR2  = mm + 8;
  unsigned* mmP2  = mm + 10;
  unsigned* mmF0  = mm + 12;
  unsigned* mmRF0 = mm + 14;
  unsigned* mmF1  = mm + 16;
  unsigned* mmRF1 = mm + 18;
  unsigned* mmLG  = mm + 20;

  const int T = 256;
  auto gemm_blocks = [](int M, int N) {
    long long waves = (long long)((M + 15) / 16) * ((N + 15) / 16);
    return (int)((waves * 32 + 255) / 256);
  };
  auto ew_blocks = [](int n) { int b = (n + 255) / 256; return b > 2048 ? 2048 : b; };

  init_mm_kernel<<<1, 32, 0, stream>>>(mm, 11);

  // conv1: [B,1,32,32] -> [B,6,28,28]
  conv_wmma_kernel<1, 32, 32, 6, 28, 28><<<gemm_blocks(B * 784, 16), T, 0, stream>>>(
      x, w1, b1, A1, B, mmA1);
  quant_kernel<<<ew_blocks(N_A1), T, 0, stream>>>(A1, N_A1, mmA1, mmR1, 1, 0, 0u, tc);
  quant_kernel<<<ew_blocks(N_A1), T, 0, stream>>>(A1, N_A1, mmR1, nullptr, 0, 1, 1001u, tc);
  maxpool_kernel<<<ew_blocks(N_P1), T, 0, stream>>>(A1, P1, 6, 28, 28, N_P1, mmP1);
  quant_kernel<<<ew_blocks(N_P1), T, 0, stream>>>(P1, N_P1, mmP1, nullptr, 0, 0, 0u, tc);

  // conv2: [B,6,14,14] -> [B,16,10,10]
  conv_wmma_kernel<6, 14, 14, 16, 10, 10><<<gemm_blocks(B * 100, 16), T, 0, stream>>>(
      P1, w2, b2, A2, B, mmA2);
  quant_kernel<<<ew_blocks(N_A2), T, 0, stream>>>(A2, N_A2, mmA2, mmR2, 1, 0, 0u, tc);
  quant_kernel<<<ew_blocks(N_A2), T, 0, stream>>>(A2, N_A2, mmR2, nullptr, 0, 1, 1002u, tc);
  maxpool_kernel<<<ew_blocks(N_P2), T, 0, stream>>>(A2, P2, 16, 10, 10, N_P2, mmP2);
  quant_kernel<<<ew_blocks(N_P2), T, 0, stream>>>(P2, N_P2, mmP2, nullptr, 0, 0, 0u, tc);

  // fc: [B,400] -> [B,120]
  fc_wmma_kernel<120, 400><<<gemm_blocks(B, 120), T, 0, stream>>>(P2, fw, fb, F0, B, mmF0);
  quant_kernel<<<ew_blocks(N_F0), T, 0, stream>>>(F0, N_F0, mmF0, mmRF0, 1, 0, 0u, tc);
  quant_kernel<<<ew_blocks(N_F0), T, 0, stream>>>(F0, N_F0, mmRF0, nullptr, 0, 1, 1003u, tc);

  // fc1: [B,120] -> [B,84]
  fc_wmma_kernel<84, 120><<<gemm_blocks(B, 84), T, 0, stream>>>(F0, f1w, f1b, F1, B, mmF1);
  quant_kernel<<<ew_blocks(N_F1), T, 0, stream>>>(F1, N_F1, mmF1, mmRF1, 1, 0, 0u, tc);
  quant_kernel<<<ew_blocks(N_F1), T, 0, stream>>>(F1, N_F1, mmRF1, nullptr, 0, 1, 1004u, tc);

  // fc2: [B,84] -> [B,10], final quant0, log_softmax
  fc_wmma_kernel<10, 84><<<gemm_blocks(B, 10), T, 0, stream>>>(F1, f2w, f2b, LG, B, mmLG);
  quant_kernel<<<ew_blocks(N_LG), T, 0, stream>>>(LG, N_LG, mmLG, nullptr, 0, 0, 0u, tc);
  logsoftmax_kernel<<<(B + T - 1) / T, T, 0, stream>>>(LG, out, B, 10);
}
